// Decoder_66065186947373
// MI455X (gfx1250) — matrix operand order repaired and checked by reading, not compiled, not run on hardware
//
#include <hip/hip_runtime.h>
#include <math.h>

// ---------------------------------------------------------------------------
// Types for CDNA5 WMMA (wave32, v_wmma_f32_16x16x32_bf16)
// ---------------------------------------------------------------------------
typedef __bf16 bf16_t;
typedef __attribute__((ext_vector_type(16))) __bf16 v16bf;
typedef __attribute__((ext_vector_type(8)))  __bf16 v8bf;
typedef __attribute__((ext_vector_type(4)))  __bf16 v4bf;
typedef __attribute__((ext_vector_type(8)))  float  v8f;
typedef __attribute__((ext_vector_type(4)))  float  v4f;

#define BM 128
#define BN 128
#define BK 32

// ---------------------------------------------------------------------------
// C[M,N] = A[M,K] @ W[K,N] (+bias) (+= existing C if accumulate)
// fp32 in global memory, bf16 in LDS / WMMA, f32 accumulation.
// Block: 256 threads = 8 waves. Block tile 128x128, K-step 32, double-buffered.
// Wave tile: 32(M) x 64(N) = 2x4 WMMA 16x16 tiles.
// Requires: M%128==0, N%128==0, K%32==0 (true for every GEMM in this model).
// ---------------------------------------------------------------------------
__global__ __launch_bounds__(256, 2)
void gemm_bf16_wmma(float* __restrict__ C,
                    const float* __restrict__ A,
                    const float* __restrict__ W,
                    const float* __restrict__ bias,
                    int M, int N, int K, int accumulate)
{
    __shared__ __align__(128) bf16_t As[2][BM * BK];   // [m][k] row-major
    __shared__ __align__(128) bf16_t Bs[2][BN * BK];   // [n][k] (K-transposed)

    const int tid  = threadIdx.x;
    const int wave = tid >> 5;
    const int lane = tid & 31;
    const int lh   = lane >> 4;       // lane half (0/1)
    const int l16  = lane & 15;
    const int wm   = wave & 3;        // 4 waves in M  -> 32-row slabs
    const int wn   = wave >> 2;       // 2 waves in N  -> 64-col slabs

    const int n0 = blockIdx.x * BN;
    const int m0 = blockIdx.y * BM;

    // global->LDS staging assignments (256 threads, 4096 elems per tile)
    const int ar  = tid >> 1;           // A row in tile (0..127)
    const int ac  = (tid & 1) * 16;     // A col in tile (0/16)
    const int wkb = (tid & 7) * 4;      // W k sub-block (0..28)
    const int wnb = (tid >> 3) * 4;     // W n sub-block (0..124)

    const int nkt = K / BK;

    v4f areg[4];
    v4f wreg[4];

    auto loadg = [&](int kt) {
        const float* ap = A + (size_t)(m0 + ar) * K + kt * BK + ac;
#pragma unroll
        for (int i = 0; i < 4; ++i) areg[i] = *(const v4f*)(ap + i * 4);
        const float* wp = W + (size_t)(kt * BK + wkb) * N + n0 + wnb;
#pragma unroll
        for (int i = 0; i < 4; ++i) wreg[i] = *(const v4f*)(wp + (size_t)i * N);
    };

    auto stlds = [&](int buf) {
        // A: 16 contiguous bf16 -> two 16B ds_writes
        bf16_t* as = &As[buf][ar * BK + ac];
        v8bf p0, p1;
#pragma unroll
        for (int j = 0; j < 8; ++j) {
            p0[j] = (bf16_t)areg[j >> 2][j & 3];
            p1[j] = (bf16_t)areg[2 + (j >> 2)][j & 3];
        }
        *(v8bf*)(as)     = p0;
        *(v8bf*)(as + 8) = p1;
        // W: transpose 4x4 micro-block into Bs[n][k] -> four 8B ds_writes
#pragma unroll
        for (int j = 0; j < 4; ++j) {
            v4bf t;
#pragma unroll
            for (int i = 0; i < 4; ++i) t[i] = (bf16_t)wreg[i][j];
            *(v4bf*)(&Bs[buf][(wnb + j) * BK + wkb]) = t;
        }
    };

    v8f acc[2][4];
#pragma unroll
    for (int im = 0; im < 2; ++im)
#pragma unroll
        for (int in = 0; in < 4; ++in)
#pragma unroll
            for (int r = 0; r < 8; ++r) acc[im][in][r] = 0.0f;

    loadg(0);
    stlds(0);
    __syncthreads();

    for (int kt = 0; kt < nkt; ++kt) {
        const int cur = kt & 1;
        if (kt + 1 < nkt) loadg(kt + 1);   // prefetch next K tile into VGPRs

        // A fragments: ISA 16-bit A 16x32 layout.
        // lane half 0 holds K {0..7,16..23}, half 1 holds K {8..15,24..31}.
        v16bf afrag[2];
#pragma unroll
        for (int im = 0; im < 2; ++im) {
            const bf16_t* ab = &As[cur][(wm * 32 + im * 16 + l16) * BK];
            v8bf alo = *(const v8bf*)(ab + lh * 8);
            v8bf ahi = *(const v8bf*)(ab + 16 + lh * 8);
            afrag[im] = __builtin_shufflevector(alo, ahi,
                0, 1, 2, 3, 4, 5, 6, 7, 8, 9, 10, 11, 12, 13, 14, 15);
        }
        // B fragments: lane half 0 holds K {0..7,16..23}, half 1 holds K {8..15,24..31}
        // for column N = l16; read from the K-transposed LDS image.
#pragma unroll
        for (int in = 0; in < 4; ++in) {
            const bf16_t* bb = &Bs[cur][(wn * 64 + in * 16 + l16) * BK];
            v8bf blo = *(const v8bf*)(bb + lh * 8);
            v8bf bhi = *(const v8bf*)(bb + 16 + lh * 8);
            v16bf bfrag = __builtin_shufflevector(blo, bhi,
                0, 1, 2, 3, 4, 5, 6, 7, 8, 9, 10, 11, 12, 13, 14, 15);
#pragma unroll
            for (int im = 0; im < 2; ++im) {
                acc[im][in] = __builtin_amdgcn_wmma_f32_16x16x32_bf16(
                    false, afrag[im], false, bfrag, (short)0, acc[im][in],
                    false, false);
            }
        }

        if (kt + 1 < nkt) {
            stlds(cur ^ 1);
            __syncthreads();
        }
    }

    // Epilogue: C/D 16x16 f32 layout: VGPR r -> row (lh*8 + r), col = l16.
#pragma unroll
    for (int in = 0; in < 4; ++in) {
        const int col = n0 + wn * 64 + in * 16 + l16;
        const float bv = bias ? bias[col] : 0.0f;
#pragma unroll
        for (int im = 0; im < 2; ++im) {
#pragma unroll
            for (int r = 0; r < 8; ++r) {
                const int row = m0 + wm * 32 + im * 16 + lh * 8 + r;
                float* cp = C + (size_t)row * N + col;
                float v = acc[im][in][r] + bv;
                if (accumulate) v += *cp;
                *cp = v;
            }
        }
    }
}

// ---------------------------------------------------------------------------
// Neighborhood attention (K_NEIGH = 8). One wave32 per token; lanes split the
// d channels (CPL = d/32). Cross-lane reductions via shfl_xor; softmax over 8
// scores in registers. Bandwidth-bound; neighbors hit L2.
// ---------------------------------------------------------------------------
template <int CPL>
__global__ __launch_bounds__(256)
void na_attn(const float* __restrict__ Q, const float* __restrict__ Kx,
             const float* __restrict__ V, const float* __restrict__ rb,
             float* __restrict__ O, int Ntok, float inv_sqrt_d)
{
    const int wave  = threadIdx.x >> 5;
    const int lane  = threadIdx.x & 31;
    const int token = blockIdx.x * 8 + wave;
    const int b = token / Ntok;
    const int n = token - b * Ntok;
    const int d = CPL * 32;

    const float* qrow = Q + (size_t)token * d;
    float q[CPL];
#pragma unroll
    for (int i = 0; i < CPL; ++i) q[i] = qrow[lane + 32 * i];

    float sc[8];
#pragma unroll
    for (int j = 0; j < 8; ++j) {
        int idx = n + j - 4;
        idx = idx < 0 ? 0 : (idx > Ntok - 1 ? Ntok - 1 : idx);
        const float* krow = Kx + ((size_t)b * Ntok + idx) * d;
        float s = 0.0f;
#pragma unroll
        for (int i = 0; i < CPL; ++i) s += q[i] * krow[lane + 32 * i];
#pragma unroll
        for (int off = 16; off > 0; off >>= 1) s += __shfl_xor(s, off, 32);
        sc[j] = (s + rb[(n & 7) * 8 + j]) * inv_sqrt_d;
    }

    float mx = sc[0];
#pragma unroll
    for (int j = 1; j < 8; ++j) mx = fmaxf(mx, sc[j]);
    float p[8], sum = 0.0f;
#pragma unroll
    for (int j = 0; j < 8; ++j) { p[j] = __expf(sc[j] - mx); sum += p[j]; }
    const float inv = 1.0f / sum;

    float o[CPL];
#pragma unroll
    for (int i = 0; i < CPL; ++i) o[i] = 0.0f;
#pragma unroll
    for (int j = 0; j < 8; ++j) {
        int idx = n + j - 4;
        idx = idx < 0 ? 0 : (idx > Ntok - 1 ? Ntok - 1 : idx);
        const float* vrow = V + ((size_t)b * Ntok + idx) * d;
        const float pj = p[j] * inv;
#pragma unroll
        for (int i = 0; i < CPL; ++i) o[i] += pj * vrow[lane + 32 * i];
    }
    float* orow = O + (size_t)token * d;
#pragma unroll
    for (int i = 0; i < CPL; ++i) orow[lane + 32 * i] = o[i];
}

// ---------------------------------------------------------------------------
// Host orchestration. Input order (setup_inputs dict order):
//   0:x 1:enc0 2:enc1 3:enc2, then per stage s (base 4+12*s):
//   W_exp, W_skip, b_skip, Wq, bq, Wk, bk, Wv, bv, Wo, bo, rel_bias
// Every intermediate tensor is exactly 8*2^20 fp32 elements (32 MiB).
// ---------------------------------------------------------------------------
extern "C" void kernel_launch(void* const* d_in, const int* in_sizes, int n_in,
                              void* d_out, int out_size, void* d_ws, size_t ws_size,
                              hipStream_t stream)
{
    (void)in_sizes; (void)n_in; (void)out_size; (void)ws_size;
    const int B = 8;
    const float* x = (const float*)d_in[0];
    const float* enc[3] = {(const float*)d_in[1], (const float*)d_in[2],
                           (const float*)d_in[3]};

    const size_t BUFE = (size_t)8 * 1024 * 1024;  // 8,388,608 floats = 32 MiB
    float* ws = (float*)d_ws;
    float* Y  = ws + 0 * BUFE;   // expand output / attention output (reused)
    float* Z  = ws + 1 * BUFE;   // post-skip activations
    float* Qb = ws + 2 * BUFE;
    float* Kb = ws + 3 * BUFE;
    float* Vb = ws + 4 * BUFE;

    const float* sin_ = x;
    int Nin = 1024;
    const dim3 blk(256);

    for (int s = 0; s < 3; ++s) {
        const int in_ch = 1024 >> s;      // 1024, 512, 256
        const int d     = in_ch >> 1;     //  512, 256, 128
        const int Nout  = 2 * Nin;
        const int M1 = B * Nin;
        const int M2 = B * Nout;

        const float* const* P = (const float* const*)(d_in + 4 + s * 12);
        const float* W_exp  = P[0];
        const float* W_skip = P[1];
        const float* b_skip = P[2];
        const float* Wq = P[3]; const float* bq = P[4];
        const float* Wk = P[5]; const float* bk = P[6];
        const float* Wv = P[7]; const float* bv = P[8];
        const float* Wo = P[9]; const float* bo = P[10];
        const float* rb = P[11];

        float* out = (float*)d_out;  // d_out doubles as the stage ping buffer

        // 1) patch expand: Y = sin_ @ W_exp   (reshape is a reinterpret)
        gemm_bf16_wmma<<<dim3(in_ch / BN, M1 / BM), blk, 0, stream>>>(
            Y, sin_, W_exp, nullptr, M1, in_ch, in_ch, 0);
        // 2) concat + skip: Z = Yr @ Ws[0:d] + b ; Z += enc @ Ws[d:2d]
        gemm_bf16_wmma<<<dim3(d / BN, M2 / BM), blk, 0, stream>>>(
            Z, Y, W_skip, b_skip, M2, d, d, 0);
        gemm_bf16_wmma<<<dim3(d / BN, M2 / BM), blk, 0, stream>>>(
            Z, enc[s], W_skip + (size_t)d * d, nullptr, M2, d, d, 1);
        // 3) Q, K, V projections
        gemm_bf16_wmma<<<dim3(d / BN, M2 / BM), blk, 0, stream>>>(
            Qb, Z, Wq, bq, M2, d, d, 0);
        gemm_bf16_wmma<<<dim3(d / BN, M2 / BM), blk, 0, stream>>>(
            Kb, Z, Wk, bk, M2, d, d, 0);
        gemm_bf16_wmma<<<dim3(d / BN, M2 / BM), blk, 0, stream>>>(
            Vb, Z, Wv, bv, M2, d, d, 0);
        // 4) neighborhood attention -> Y (reused)
        const float isd = 1.0f / sqrtf((float)d);
        const int nblk = M2 / 8;
        if (d == 512)
            na_attn<16><<<nblk, blk, 0, stream>>>(Qb, Kb, Vb, rb, Y, Nout, isd);
        else if (d == 256)
            na_attn<8><<<nblk, blk, 0, stream>>>(Qb, Kb, Vb, rb, Y, Nout, isd);
        else
            na_attn<4><<<nblk, blk, 0, stream>>>(Qb, Kb, Vb, rb, Y, Nout, isd);
        // 5) output projection
        gemm_bf16_wmma<<<dim3(d / BN, M2 / BM), blk, 0, stream>>>(
            out, Y, Wo, bo, M2, d, d, 0);

        sin_ = out;
        Nin = Nout;
    }
}
